// InPlaceTTT_64957085384845
// MI455X (gfx1250) — compile-verified, matrix operand order checked by reading, not run on hardware
//
#include <hip/hip_runtime.h>
#include <hip/hip_bf16.h>

#define DEVI __device__ __forceinline__

typedef __attribute__((ext_vector_type(8)))  unsigned short v8us;
typedef __attribute__((ext_vector_type(16))) unsigned short v16us;
typedef __attribute__((ext_vector_type(16))) __bf16         v16bf;
typedef __attribute__((ext_vector_type(8)))  float          v8f;

union V16cast { v16us u; v16bf b; };

static constexpr int Ndim = 4096;   // B*T
static constexpr int Ddim = 2048;
static constexpr int Rdim = 32;
static constexpr int Vdim = 32000;

// ---------- helpers ----------
DEVI unsigned short f2us(float f) {            // f32 -> bf16 bits (truncate)
  union { float f; unsigned u; } v; v.f = f;
  return (unsigned short)(v.u >> 16);
}

DEVI v8f v8f_zero() { v8f z = {0.f,0.f,0.f,0.f,0.f,0.f,0.f,0.f}; return z; }

// 16x32 bf16 operand from LDS, ISA lane layout.
// p = row base (lane row & k-offset applied), kb = (lane>=16)?8:0
DEVI v16bf lds_amat(const unsigned short* p, int kb) {
  v8us lo = *(const v8us*)(p + kb);
  v8us hi = *(const v8us*)(p + kb + 16);
  V16cast c;
  c.u = __builtin_shufflevector(lo, hi, 0,1,2,3,4,5,6,7,8,9,10,11,12,13,14,15);
  return c.b;
}

// same, but straight from global bf16 memory (p already includes kb)
DEVI v16bf glb_amat(const unsigned short* p) {
  v8us lo = *(const v8us*)(p);
  v8us hi = *(const v8us*)(p + 16);
  V16cast c;
  c.u = __builtin_shufflevector(lo, hi, 0,1,2,3,4,5,6,7,8,9,10,11,12,13,14,15);
  return c.b;
}

DEVI v8f wmma_bf16(v16bf a, v16bf b, v8f c) {
  return __builtin_amdgcn_wmma_f32_16x16x32_bf16(false, a, false, b, (short)0, c,
                                                 false, false);
}

// ---------- K0: partial column sums of hidden (32 row-chunks) ----------
__global__ __launch_bounds__(256)
void colmean_kernel(const float* __restrict__ h, float* __restrict__ part) {
  int d = blockIdx.x * 256 + threadIdx.x;   // 8 blocks in x
  int c = blockIdx.y;                        // 32 chunks of 128 rows
  float s = 0.f;
  for (int n = c * 128; n < c * 128 + 128; ++n) s += h[(size_t)n * Ddim + d];
  part[(size_t)c * Ddim + d] = s;
}

// ---------- K1: alpha = sigmoid(mean(h) . gW + gb), step = exp(lss) --------
__global__ __launch_bounds__(256)
void scalars_kernel(const float* __restrict__ part, const float* __restrict__ gw,
                    const float* __restrict__ gb, const float* __restrict__ lss,
                    float* __restrict__ scal) {
  __shared__ float red[256];
  int tid = threadIdx.x;
  float s = 0.f;
  for (int d = tid; d < Ddim; d += 256) {
    float p = 0.f;
    for (int c = 0; c < 32; ++c) p += part[(size_t)c * Ddim + d];
    s += p * gw[d];
  }
  red[tid] = s; __syncthreads();
  for (int o = 128; o > 0; o >>= 1) {
    if (tid < o) red[tid] += red[tid + o];
    __syncthreads();
  }
  if (tid == 0) {
    float x = red[0] * (1.0f / (float)Ndim) + gb[0];
    scal[0] = 1.0f / (1.0f + __expf(-x));   // alpha
    scal[1] = __expf(lss[0]);               // step
  }
}

// ---------- K1b: Wb = bf16(W)  (one-time; streamed operand becomes L2-resident)
__global__ __launch_bounds__(256)
void wconv_kernel(const float* __restrict__ W, unsigned short* __restrict__ Wb) {
  size_t i = ((size_t)blockIdx.x * 256 + threadIdx.x) * 4;
  float4 x = *(const float4*)(W + i);
  ushort4 u;
  u.x = f2us(x.x); u.y = f2us(x.y); u.z = f2us(x.z); u.w = f2us(x.w);
  *(ushort4*)(Wb + i) = u;
}

// ---------- K2: generic NT GEMM  C[M,Nc] = s*(A[M,K] . B[Nc,K]^T) + base ----
__global__ __launch_bounds__(256)
void gemm_nt_kernel(const float* __restrict__ A, const float* __restrict__ B,
                    const float* __restrict__ base, const float* __restrict__ scaleP,
                    float* __restrict__ C, unsigned short* __restrict__ Cb,
                    int M, int Nc, int K) {
  __shared__ unsigned short As[64 * 40];
  __shared__ unsigned short Bs[64 * 40];
  int tid = threadIdx.x, lane = tid & 31, wave = tid >> 5;
  int hi = lane >> 4, ln = lane & 15, kb = hi ? 8 : 0;
  int m0 = blockIdx.x * 64, nb0 = blockIdx.y * 64;
  int mo = (wave >> 1) * 16, no = (wave & 1) * 32;
  int rowL = tid >> 2, cg = (tid & 3) * 8;

  v8f acc0 = v8f_zero(), acc1 = v8f_zero();

  for (int k0 = 0; k0 < K; k0 += 32) {
    { // stage A tile (64x32)
      const float* src = A + (size_t)(m0 + rowL) * K + k0 + cg;
      unsigned short* d = As + rowL * 40 + cg;
      float4 x = *(const float4*)src; float4 y = *(const float4*)(src + 4);
      d[0]=f2us(x.x); d[1]=f2us(x.y); d[2]=f2us(x.z); d[3]=f2us(x.w);
      d[4]=f2us(y.x); d[5]=f2us(y.y); d[6]=f2us(y.z); d[7]=f2us(y.w);
    }
    { // stage B tile (64x32), zero-pad rows >= Nc
      unsigned short* d = Bs + rowL * 40 + cg;
      int bn = nb0 + rowL;
      if (bn < Nc) {
        const float* src = B + (size_t)bn * K + k0 + cg;
        float4 x = *(const float4*)src; float4 y = *(const float4*)(src + 4);
        d[0]=f2us(x.x); d[1]=f2us(x.y); d[2]=f2us(x.z); d[3]=f2us(x.w);
        d[4]=f2us(y.x); d[5]=f2us(y.y); d[6]=f2us(y.z); d[7]=f2us(y.w);
      } else {
        for (int j = 0; j < 8; ++j) d[j] = 0;
      }
    }
    __syncthreads();
    v16bf a  = lds_amat(As + (mo + ln) * 40, kb);
    v16bf b0 = lds_amat(Bs + (no + ln) * 40, kb);
    v16bf b1 = lds_amat(Bs + (no + 16 + ln) * 40, kb);
    acc0 = wmma_bf16(a, b0, acc0);
    acc1 = wmma_bf16(a, b1, acc1);
    __syncthreads();
  }

  float s = scaleP ? scaleP[0] : 1.0f;
#pragma unroll
  for (int t = 0; t < 2; ++t) {
    v8f ac = t ? acc1 : acc0;
#pragma unroll
    for (int j = 0; j < 8; ++j) {
      int row = m0 + mo + 8 * hi + j;
      int col = nb0 + no + 16 * t + ln;
      if (col < Nc) {
        size_t idx = (size_t)row * Nc + col;
        float v = s * ac[j];
        if (base) v += base[idx];
        C[idx] = v;
        if (Cb) Cb[idx] = f2us(v);
      }
    }
  }
}

// ---------- K3: fused logits / online-softmax / grad ------------------------
// G = (softmax(o W^T) W - W[tgt]) / N, single pass over V, G in registers.
// o-operands live in VGPRs for the whole V-loop; W vtile double-buffered in LDS.
static constexpr int OSTR = 2056;   // ushorts per staged row (2048 + 8 pad)
static constexpr int PSTR = 40;     // ushorts per staged P row
static constexpr int K3_SMEM = 2 * 32 * OSTR * 2  // WtS[2] (bf16 W vtiles)
                             + 8 * 32 * 32 * 4    // lpart
                             + 32 * 32 * 4        // ltile
                             + 32 * PSTR * 2      // pS
                             + 3 * 32 * 4         // rowM,rowS,rowF
                             + 16;                // sFlag (+pad)

__global__ __launch_bounds__(256)
void fused_ce_grad_kernel(const unsigned short* __restrict__ ob,
                          const unsigned short* __restrict__ Wb,
                          const int* __restrict__ tgt, float* __restrict__ G) {
  extern __shared__ char smem[];
  unsigned short* WtS  = (unsigned short*)smem;            // 2 buffers
  float*          lpart= (float*)(WtS + 2 * 32 * OSTR);
  float*          ltile= lpart + 8 * 32 * 32;
  unsigned short* pS   = (unsigned short*)(ltile + 32 * 32);
  float*          rowM = (float*)(pS + 32 * PSTR);
  float*          rowS = rowM + 32;
  float*          rowF = rowS + 32;
  int*            sFlag= (int*)(rowF + 32);

  int tid = threadIdx.x, lane = tid & 31, wave = tid >> 5;
  int hi = lane >> 4, ln = lane & 15, kb = hi ? 8 : 0;
  int n0 = blockIdx.x * 32;
  int rS = tid >> 3, cS = (tid & 7) * 256;     // cooperative staging mapping

  const int kslice = wave * 256;               // this wave's K range of D

  // hoist this wave's o-operands (32 rows x 256 K) into registers: invariant
  v16bf aregs[16];
#pragma unroll
  for (int ks = 0; ks < 8; ++ks) {
    aregs[2 * ks]     = glb_amat(ob + (size_t)(n0 + ln)      * Ddim + kslice + ks * 32 + kb);
    aregs[2 * ks + 1] = glb_amat(ob + (size_t)(n0 + 16 + ln) * Ddim + kslice + ks * 32 + kb);
  }

  { // prologue: stage W vtile 0 into buffer 0
    const unsigned short* src = Wb + (size_t)rS * Ddim + cS;
    unsigned short* dst = WtS + rS * OSTR + cS;
#pragma unroll
    for (int j = 0; j < 256; j += 8)
      *(v8us*)(dst + j) = *(const v8us*)(src + j);
  }
  if (tid < 32) { rowM[tid] = -1e30f; rowS[tid] = 0.f; }

  v8f acc[2][16];
#pragma unroll
  for (int mt = 0; mt < 2; ++mt)
#pragma unroll
    for (int nt = 0; nt < 16; ++nt) acc[mt][nt] = v8f_zero();
  __syncthreads();

  for (int v0 = 0; v0 < Vdim; v0 += 32) {
    int t = v0 >> 5;
    unsigned short* Wcur = WtS + (t & 1)        * 32 * OSTR;
    unsigned short* Wnxt = WtS + ((t + 1) & 1)  * 32 * OSTR;

    // ---- issue stage of vtile t+1 into the other buffer (overlaps A..C)
    if (v0 + 32 < Vdim) {
      const unsigned short* src = Wb + (size_t)(v0 + 32 + rS) * Ddim + cS;
      unsigned short* dst = Wnxt + rS * OSTR + cS;
#pragma unroll
      for (int j = 0; j < 256; j += 8)
        *(v8us*)(dst + j) = *(const v8us*)(src + j);
    }
    if (v0 + 64 < Vdim)
      __builtin_prefetch(Wb + (size_t)(v0 + 64 + rS) * Ddim + cS, 0, 1);

    // ---- phase A: partial logits (32 rows x 32 vocab), K split across waves
    v8f la[2][2];
    la[0][0]=v8f_zero(); la[0][1]=v8f_zero(); la[1][0]=v8f_zero(); la[1][1]=v8f_zero();
#pragma unroll
    for (int ks = 0; ks < 8; ++ks) {
      int kofs = kslice + ks * 32;
      v16bf b0 = lds_amat(Wcur + (ln)      * OSTR + kofs, kb);
      v16bf b1 = lds_amat(Wcur + (16 + ln) * OSTR + kofs, kb);
      la[0][0] = wmma_bf16(aregs[2 * ks],     b0, la[0][0]);
      la[0][1] = wmma_bf16(aregs[2 * ks],     b1, la[0][1]);
      la[1][0] = wmma_bf16(aregs[2 * ks + 1], b0, la[1][0]);
      la[1][1] = wmma_bf16(aregs[2 * ks + 1], b1, la[1][1]);
    }
#pragma unroll
    for (int mt = 0; mt < 2; ++mt)
#pragma unroll
      for (int vt = 0; vt < 2; ++vt)
#pragma unroll
        for (int j = 0; j < 8; ++j)
          lpart[wave * 1024 + (16 * mt + 8 * hi + j) * 32 + 16 * vt + ln] = la[mt][vt][j];
    __syncthreads();

    // ---- phase B1: cross-wave reduce (256 threads, 4 cells each)
    {
      int cell = tid * 4;
#pragma unroll
      for (int q = 0; q < 4; ++q) {
        int idx = cell + q;
        float s = 0.f;
#pragma unroll
        for (int w2 = 0; w2 < 8; ++w2) s += lpart[w2 * 1024 + idx];
        ltile[idx] = s;
      }
    }
    __syncthreads();
    // ---- phase B2: online softmax (one wave, one row per thread)
    if (tid < 32) {
      float tm = -1e30f;
      for (int c = 0; c < 32; ++c) tm = fmaxf(tm, ltile[tid * 32 + c]);
      float oldM = rowM[tid];
      float newM = fmaxf(oldM, tm);
      float f = __expf(oldM - newM);
      float s = 0.f;
      for (int c = 0; c < 32; ++c) {
        float p = __expf(ltile[tid * 32 + c] - newM);
        s += p;
        pS[tid * PSTR + c] = f2us(p);
      }
      rowF[tid] = f;
      rowS[tid] = rowS[tid] * f + s;
      rowM[tid] = newM;
      unsigned long long bal = __ballot(f != 1.0f);
      if (tid == 0) sFlag[0] = (bal != 0ull) ? 1 : 0;
    }
    __syncthreads();

    // ---- phase C: (conditional) rescale + accumulate P @ W into registers
    if (sFlag[0]) {
      float fr[2][8];
#pragma unroll
      for (int mt = 0; mt < 2; ++mt)
#pragma unroll
        for (int j = 0; j < 8; ++j) fr[mt][j] = rowF[16 * mt + 8 * hi + j];
#pragma unroll
      for (int nt = 0; nt < 16; ++nt)
#pragma unroll
        for (int mt = 0; mt < 2; ++mt)
#pragma unroll
          for (int j = 0; j < 8; ++j) acc[mt][nt][j] *= fr[mt][j];
    }
    v16bf p0 = lds_amat(pS + (ln)      * PSTR, kb);
    v16bf p1 = lds_amat(pS + (16 + ln) * PSTR, kb);
#pragma unroll
    for (int nt = 0; nt < 16; ++nt) {
      int dcol = wave * 256 + nt * 16 + ln;
      // transposed gather from the staged W tile: BT[d][v] = Wcur[v][d]
      v16us w;
#pragma unroll
      for (int e = 0; e < 8; ++e) w[e]     = Wcur[(size_t)(kb + e) * OSTR + dcol];
#pragma unroll
      for (int e = 0; e < 8; ++e) w[8 + e] = Wcur[(size_t)(kb + 16 + e) * OSTR + dcol];
      V16cast cc; cc.u = w;
      v16bf bt = cc.b;
#pragma unroll
      for (int mt = 0; mt < 2; ++mt)
        acc[mt][nt] = wmma_bf16(mt ? p1 : p0, bt, acc[mt][nt]);
    }
    __syncthreads();
  }

  // ---- epilogue: G = (acc/rowS - W[tgt]) / N
  const float invN = 1.0f / (float)Ndim;
#pragma unroll
  for (int mt = 0; mt < 2; ++mt) {
#pragma unroll
    for (int j = 0; j < 8; ++j) {
      int row = 16 * mt + 8 * hi + j;
      int n = n0 + row;
      float inv = 1.0f / rowS[row];
      const unsigned short* wt = Wb + (size_t)tgt[n] * Ddim;
#pragma unroll
      for (int nt = 0; nt < 16; ++nt) {
        int d = wave * 256 + nt * 16 + ln;
        union { unsigned u; float f; } wv; wv.u = ((unsigned)wt[d]) << 16;
        G[(size_t)n * Ddim + d] = (acc[mt][nt][j] * inv - wv.f) * invN;
      }
    }
  }
}

// ---------- K4: GA = alpha * (G . delta_A)   [N,R] ----------
__global__ __launch_bounds__(256)
void ga_row_kernel(const float* __restrict__ G, const float* __restrict__ dA,
                   const float* __restrict__ scal, float* __restrict__ GA) {
  int n = blockIdx.x * 8 + (threadIdx.x >> 5);
  int r = threadIdx.x & 31;
  const float* g = G + (size_t)n * Ddim;
  float acc = 0.f;
  for (int d = 0; d < Ddim; ++d) acc += g[d] * dA[d * Rdim + r];
  GA[(size_t)n * Rdim + r] = scal[0] * acc;
}

// ---------- K5: out = alpha*base - step * (big^T . small) ------------------
// big [N,X] (contract over rows), small [N,R].  trans=0 -> out[x,r]; 1 -> out[r,x]
__global__ __launch_bounds__(256)
void tn_small_kernel(const float* __restrict__ big, const float* __restrict__ small,
                     const float* __restrict__ base, const float* __restrict__ scal,
                     float* __restrict__ out, int X, int trans) {
  __shared__ float sm[128 * 32];
  int tid = threadIdx.x;
  int xg = tid & 63, rg = tid >> 6;
  int x = blockIdx.x * 64 + xg;
  int r0 = rg * 8;
  float acc[8];
#pragma unroll
  for (int j = 0; j < 8; ++j) acc[j] = 0.f;
  for (int nc = 0; nc < Ndim; nc += 128) {
    for (int i = tid; i < 128 * 32; i += 256)
      sm[i] = small[(size_t)(nc + (i >> 5)) * Rdim + (i & 31)];
    __syncthreads();
    for (int nn = 0; nn < 128; ++nn) {
      float gv = big[(size_t)(nc + nn) * X + x];
#pragma unroll
      for (int j = 0; j < 8; ++j) acc[j] += gv * sm[nn * 32 + r0 + j];
    }
    __syncthreads();
  }
  float alpha = scal[0], step = scal[1];
#pragma unroll
  for (int j = 0; j < 8; ++j) {
    int r = r0 + j;
    size_t idx = trans ? ((size_t)r * X + x) : ((size_t)x * Rdim + r);
    out[idx] = alpha * base[idx] - step * acc[j];
  }
}

// ---------- launch ----------
extern "C" void kernel_launch(void* const* d_in, const int* in_sizes, int n_in,
                              void* d_out, int out_size, void* d_ws, size_t ws_size,
                              hipStream_t stream) {
  (void)in_sizes; (void)n_in; (void)out_size; (void)ws_size;
  const float* hidden  = (const float*)d_in[0];
  const int*   targets = (const int*)d_in[1];
  const float* W       = (const float*)d_in[2];
  const float* dA      = (const float*)d_in[3];
  const float* dB      = (const float*)d_in[4];
  const float* gw      = (const float*)d_in[5];
  const float* gb      = (const float*)d_in[6];
  const float* lss     = (const float*)d_in[7];

  float* o_out = (float*)d_out;                       // [N,D]
  float* Anew  = o_out + (size_t)Ndim * Ddim;         // [D,R]
  float* Bnew  = Anew + (size_t)Ddim * Rdim;          // [R,D]

  float* ws   = (float*)d_ws;
  float* part = ws;                                    // [32,D] partial col sums
  float* scal = ws + 32 * Ddim;                        // alpha, step
  float* z    = scal + 64;                             // [N,R]
  float* GA   = z + (size_t)Ndim * Rdim;               // [N,R]
  float* G    = GA + (size_t)Ndim * Rdim;              // [N,D] f32
  unsigned short* Wb = (unsigned short*)(G + (size_t)Ndim * Ddim); // [V,D] bf16
  unsigned short* ob = Wb + (size_t)Vdim * Ddim;                   // [N,D] bf16

  colmean_kernel<<<dim3(Ddim / 256, 32), 256, 0, stream>>>(hidden, part);
  scalars_kernel<<<1, 256, 0, stream>>>(part, gw, gb, lss, scal);
  wconv_kernel<<<(Vdim / 4) * (Ddim / 256), 256, 0, stream>>>(W, Wb);

  // z = alpha * (h . dB^T)
  gemm_nt_kernel<<<dim3(Ndim / 64, 1), 256, 0, stream>>>(
      hidden, dB, nullptr, scal, z, nullptr, Ndim, Rdim, Ddim);
  // o = h + alpha * (z . dA^T)   (also emits bf16 copy for the fused pass)
  gemm_nt_kernel<<<dim3(Ndim / 64, Ddim / 64), 256, 0, stream>>>(
      z, dA, hidden, scal, o_out, ob, Ndim, Ddim, Rdim);

  hipFuncSetAttribute(reinterpret_cast<const void*>(fused_ce_grad_kernel),
                      hipFuncAttributeMaxDynamicSharedMemorySize, K3_SMEM);
  fused_ce_grad_kernel<<<Ndim / 32, 256, K3_SMEM, stream>>>(ob, Wb, targets, G);

  ga_row_kernel<<<Ndim / 8, 256, 0, stream>>>(G, dA, scal, GA);
  // A_new = alpha*dA - step * (G^T . z)
  tn_small_kernel<<<Ddim / 64, 256, 0, stream>>>(G, z, dA, scal, Anew, Ddim, 0);
  // B_new = alpha*dB - step * (GA^T . h)
  tn_small_kernel<<<Ddim / 64, 256, 0, stream>>>(hidden, GA, dB, scal, Bnew, Ddim, 1);
}
